// FullLinearAttention_70635032150391
// MI455X (gfx1250) — compile-verified, matrix operand order checked by reading, not run on hardware
//
#include <hip/hip_runtime.h>
#include <hip/hip_bf16.h>

// ---------- types ----------
typedef __attribute__((ext_vector_type(16))) _Float16 v16h;
typedef __attribute__((ext_vector_type(8)))  _Float16 v8h;
typedef __attribute__((ext_vector_type(4)))  _Float16 v4h;
typedef __attribute__((ext_vector_type(8)))  float    v8f;
typedef __attribute__((ext_vector_type(4)))  float    v4f;

// ---------- problem constants ----------
#define BB   4
#define SS   8192
#define DD   1024
#define HH   16
#define HD   64
#define BH   (BB*HH)        // 64
#define MROWS (BB*SS)       // 32768
#define N3   (3*DD)         // 3072

#define LDA  40             // padded LDS stride (f16): 80B rows, 16B-aligned
#define LDKV 72             // padded LDS stride (f16) for kv tile: 144B rows, 16B-aligned

// ---------- fragment helpers (ISA 16-bit A/B layout, wave32) ----------
// lane<16 : row/col = lane,    K runs {0..7, 16..23}
// lane>=16: row/col = lane-16, K runs {8..15, 24..31}
__device__ __forceinline__ v16h cat16(v8h lo, v8h hi) {
  return __builtin_shufflevector(lo, hi, 0,1,2,3,4,5,6,7,8,9,10,11,12,13,14,15);
}
__device__ __forceinline__ v16h frag_ld(const _Float16* rowbase, int kh) {
  v8h lo = *(const v8h*)(rowbase + 8*kh);
  v8h hi = *(const v8h*)(rowbase + 16 + 8*kh);
  return cat16(lo, hi);
}
__device__ __forceinline__ v8f wmma16(v16h a, v16h b, v8f c) {
  return __builtin_amdgcn_wmma_f32_16x16x32_f16(false, a, false, b, (short)0, c, false, false);
}

// ============================================================
// Stage 0: zero kv accumulator
// ============================================================
__global__ void la_init_kv(float* __restrict__ kv) {
  kv[blockIdx.x * 256 + threadIdx.x] = 0.0f;   // BH*64*64 = 262144 floats
}

// ============================================================
// Stage 1: qkv = x @ w_qkv^T  (f32 in, f16 WMMA, f32 acc)
// q -> [B,H,S,64] (elu+1),  k -> [B,H,64,S] (elu+1, transposed), v -> [B,H,64,S]
// block tile 128(M) x 128(N), 8 waves -> 32x64 per wave (2x4 WMMA tiles), K step 32
// ============================================================
__global__ __launch_bounds__(256) void la_qkv_gemm(
    const float* __restrict__ x, const float* __restrict__ wqkv,
    _Float16* __restrict__ qf, _Float16* __restrict__ kT, _Float16* __restrict__ vT)
{
  __shared__ _Float16 As[128 * LDA];
  __shared__ _Float16 Bs[128 * LDA];
  const int tid = threadIdx.x, lane = tid & 31, w = tid >> 5;
  const int wm = w & 3, wn = w >> 2;          // wave tile: rows wm*32, cols wn*64
  const int mbase = blockIdx.y * 128, nbase = blockIdx.x * 128;
  const int ml = lane & 15, kh = lane >> 4;

  v8f acc[2][4] = {};

  for (int k0 = 0; k0 < DD; k0 += 32) {
    __syncthreads();
    // stage A: 128x32 f32 -> f16 (1024 float4, 4 consecutive per thread)
#pragma unroll
    for (int i = 0; i < 4; ++i) {
      int idx = tid * 4 + i, r = idx >> 3, c4 = idx & 7;
      const float* gp = x + (size_t)(mbase + r) * DD + k0 + c4 * 4;
      if (i == 0) __builtin_prefetch(gp + 32, 0, 1);   // speculative: next K tile
      v4f d = *(const v4f*)gp;
      v4h h = { (_Float16)d[0], (_Float16)d[1], (_Float16)d[2], (_Float16)d[3] };
      *(v4h*)(As + r * LDA + c4 * 4) = h;
    }
    // stage B: 128x32 f32 -> f16, Bs[n][k] = W[n][k]
#pragma unroll
    for (int i = 0; i < 4; ++i) {
      int idx = tid * 4 + i, r = idx >> 3, c4 = idx & 7;
      const float* gp = wqkv + (size_t)(nbase + r) * DD + k0 + c4 * 4;
      if (i == 0) __builtin_prefetch(gp + 32, 0, 1);
      v4f d = *(const v4f*)gp;
      v4h h = { (_Float16)d[0], (_Float16)d[1], (_Float16)d[2], (_Float16)d[3] };
      *(v4h*)(Bs + r * LDA + c4 * 4) = h;
    }
    __syncthreads();
    v16h a[2], b[4];
#pragma unroll
    for (int mi = 0; mi < 2; ++mi) a[mi] = frag_ld(As + (wm*32 + mi*16 + ml) * LDA, kh);
#pragma unroll
    for (int ni = 0; ni < 4; ++ni) b[ni] = frag_ld(Bs + (wn*64 + ni*16 + ml) * LDA, kh);
#pragma unroll
    for (int mi = 0; mi < 2; ++mi)
#pragma unroll
      for (int ni = 0; ni < 4; ++ni) acc[mi][ni] = wmma16(a[mi], b[ni], acc[mi][ni]);
  }

  // epilogue: split q/k/v, feature map, scatter
#pragma unroll
  for (int mi = 0; mi < 2; ++mi)
#pragma unroll
    for (int ni = 0; ni < 4; ++ni) {
      int n = nbase + wn*64 + ni*16 + ml;
      int part = n >> 10, rem = n & 1023;
      int h = rem >> 6, hd = rem & 63;
#pragma unroll
      for (int r = 0; r < 8; ++r) {
        int m = mbase + wm*32 + mi*16 + r + 8*kh;
        int bb = m >> 13, s = m & (SS - 1);
        float val = acc[mi][ni][r];
        if (part < 2) val = (val > 0.0f) ? (val + 1.0f) : __expf(val);  // elu(x)+1
        _Float16 hv = (_Float16)val;
        size_t bh = (size_t)(bb * HH + h);
        if (part == 0)      qf[((bh * SS + s) << 6) + hd] = hv;
        else if (part == 1) kT[((bh * HD + hd) << 13) + s] = hv;
        else                vT[((bh * HD + hd) << 13) + s] = hv;
      }
    }
}

// ============================================================
// Stage 2: kv[b,h,d,e] += sum_s kT[d,s] * vT[e,s]   (S split across blocks)
// A = k^T (M=d,K=s), B = v (K=s,N=e) read column-major from vT -> contiguous
// ============================================================
__global__ __launch_bounds__(256) void la_kv_partial(
    const _Float16* __restrict__ kT, const _Float16* __restrict__ vT,
    float* __restrict__ kv)
{
  const int bh = blockIdx.y, schunk = blockIdx.x;       // 64 chunks * 128 s
  const int tid = threadIdx.x, lane = tid & 31, w = tid >> 5;
  const int dtile = w >> 1, epair = w & 1;
  const int ml = lane & 15, kh = lane >> 4;
  const _Float16* kbase = kT + ((size_t)bh * HD + dtile*16 + ml) * SS;

  v8f acc[2] = {};
  const int s0 = schunk * 128;
#pragma unroll
  for (int kk = 0; kk < 128; kk += 32) {
    int sk = s0 + kk;
    v16h a = cat16(*(const v8h*)(kbase + sk + 8*kh), *(const v8h*)(kbase + sk + 16 + 8*kh));
#pragma unroll
    for (int ni = 0; ni < 2; ++ni) {
      const _Float16* vbase = vT + ((size_t)bh * HD + (epair*2 + ni)*16 + ml) * SS;
      v16h b = cat16(*(const v8h*)(vbase + sk + 8*kh), *(const v8h*)(vbase + sk + 16 + 8*kh));
      acc[ni] = wmma16(a, b, acc[ni]);
    }
  }
#pragma unroll
  for (int ni = 0; ni < 2; ++ni) {
    int e = (epair*2 + ni)*16 + ml;
#pragma unroll
    for (int r = 0; r < 8; ++r) {
      int d = dtile*16 + r + 8*kh;
      atomicAdd(&kv[(size_t)bh * 4096 + d * 64 + e], acc[ni][r]);
    }
  }
}

// ============================================================
// Stage 3a: k_sum[b,h,d] = sum_s kT[b,h,d,s]   (one wave per row)
// ============================================================
__global__ __launch_bounds__(256) void la_ksum(
    const _Float16* __restrict__ kT, float* __restrict__ ksum)
{
  int gw = (blockIdx.x * 256 + threadIdx.x) >> 5;   // 0..4095 = bh*64 + d
  int lane = threadIdx.x & 31;
  const _Float16* base = kT + (size_t)gw * SS;
  float acc = 0.0f;
  for (int i = 0; i < 32; ++i) {
    v8h v = *(const v8h*)(base + i * 256 + lane * 8);
#pragma unroll
    for (int j = 0; j < 8; ++j) acc += (float)v[j];
  }
#pragma unroll
  for (int off = 16; off; off >>= 1) acc += __shfl_xor(acc, off, 32);
  if (lane == 0) ksum[gw] = acc;
}

// ============================================================
// Stage 3b: kvT[b,h,e,d] = (f16) kv[b,h,d,e]   (transpose for q@kv B-fragments)
// ============================================================
__global__ void la_kv_finalize(const float* __restrict__ kv, _Float16* __restrict__ kvT) {
  int t = blockIdx.x * 256 + threadIdx.x;           // over [bh][e][d]
  int bh = t >> 12, ed = t & 4095, e = ed >> 6, d = ed & 63;
  kvT[t] = (_Float16)kv[(size_t)bh * 4096 + d * 64 + e];
}

// ============================================================
// Stage 3c: rnorm[b,h,s] = 1 / max(q . k_sum, 1e-6)
// ============================================================
__global__ void la_norm(const _Float16* __restrict__ q, const float* __restrict__ ksum,
                        float* __restrict__ rnorm) {
  int g = blockIdx.x * 256 + threadIdx.x;           // 0 .. BH*S-1
  int bh = g >> 13;
  const _Float16* qp = q + (size_t)g * HD;
  const float* ks = ksum + bh * HD;
  float dot = 0.0f;
#pragma unroll
  for (int j = 0; j < HD; ++j) dot += (float)qp[j] * ks[j];
  rnorm[g] = 1.0f / fmaxf(dot, 1e-6f);
}

// ============================================================
// Stage 4: attn[b,s,h,e] = (q[b,h,s,:] @ kv) * rnorm   -> f16 [B,S,1024]
// block: 128 s rows x 64 e cols per (b,h); kvT staged in LDS
// ============================================================
__global__ __launch_bounds__(256) void la_attn(
    const _Float16* __restrict__ q, const _Float16* __restrict__ kvT,
    const float* __restrict__ rnorm, _Float16* __restrict__ attn)
{
  __shared__ _Float16 kvs[64 * LDKV];
  __shared__ float rns[128];
  const int bh = blockIdx.y, mbase = blockIdx.x * 128;
  const int tid = threadIdx.x, lane = tid & 31, w = tid >> 5;
  const int wm = w & 3, wn = w >> 2;
  const int ml = lane & 15, kh = lane >> 4;

  { // stage kvT (64x64 f16) into padded LDS
    int e = tid >> 2, ch = tid & 3;
    const _Float16* src = kvT + (size_t)bh * 4096 + e * 64 + ch * 16;
    *(v8h*)(kvs + e * LDKV + ch * 16)     = *(const v8h*)src;
    *(v8h*)(kvs + e * LDKV + ch * 16 + 8) = *(const v8h*)(src + 8);
  }
  if (tid < 128) rns[tid] = rnorm[(size_t)bh * SS + mbase + tid];
  __syncthreads();

  v8f acc[2][2] = {};
#pragma unroll
  for (int k0 = 0; k0 < HD; k0 += 32) {
    v16h a[2], b[2];
#pragma unroll
    for (int mi = 0; mi < 2; ++mi) {
      const _Float16* qp = q + ((size_t)bh * SS + mbase + wm*32 + mi*16 + ml) * HD + k0;
      a[mi] = cat16(*(const v8h*)(qp + 8*kh), *(const v8h*)(qp + 16 + 8*kh));
    }
#pragma unroll
    for (int ni = 0; ni < 2; ++ni) b[ni] = frag_ld(kvs + (wn*32 + ni*16 + ml) * LDKV + k0, kh);
#pragma unroll
    for (int mi = 0; mi < 2; ++mi)
#pragma unroll
      for (int ni = 0; ni < 2; ++ni) acc[mi][ni] = wmma16(a[mi], b[ni], acc[mi][ni]);
  }

  const int bb = bh >> 4, hh = bh & 15;
#pragma unroll
  for (int mi = 0; mi < 2; ++mi)
#pragma unroll
    for (int ni = 0; ni < 2; ++ni) {
      int e = wn*32 + ni*16 + ml;
#pragma unroll
      for (int r = 0; r < 8; ++r) {
        int sl = wm*32 + mi*16 + r + 8*kh;
        float v = acc[mi][ni][r] * rns[sl];
        int s = mbase + sl;
        attn[(((size_t)bb * SS + s) * HH + hh) * HD + e] = (_Float16)v;
      }
    }
}

// ============================================================
// Stage 5: out = attn @ w_out^T  (f16 A, f32 weights->f16, f32 out)
// block tile 128x128, wave tile 32x64
// ============================================================
__global__ __launch_bounds__(256) void la_out_gemm(
    const _Float16* __restrict__ a16, const float* __restrict__ wout,
    float* __restrict__ out)
{
  __shared__ _Float16 As[128 * LDA];
  __shared__ _Float16 Bs[128 * LDA];
  const int tid = threadIdx.x, lane = tid & 31, w = tid >> 5;
  const int wm = w & 3, wn = w >> 2;
  const int mbase = blockIdx.y * 128, nbase = blockIdx.x * 128;
  const int ml = lane & 15, kh = lane >> 4;

  v8f acc[2][4] = {};

  for (int k0 = 0; k0 < DD; k0 += 32) {
    __syncthreads();
    // stage A: 128x32 f16 (512 x v8h, 2 per thread)
#pragma unroll
    for (int i = 0; i < 2; ++i) {
      int idx = tid * 2 + i, r = idx >> 2, ch = idx & 3;
      const _Float16* gp = a16 + (size_t)(mbase + r) * DD + k0 + ch * 8;
      if (i == 0) __builtin_prefetch(gp + 32, 0, 1);   // speculative: next K tile
      *(v8h*)(As + r * LDA + ch * 8) = *(const v8h*)gp;
    }
    // stage B: 128x32 f32 -> f16
#pragma unroll
    for (int i = 0; i < 4; ++i) {
      int idx = tid * 4 + i, r = idx >> 3, c4 = idx & 7;
      const float* gp = wout + (size_t)(nbase + r) * DD + k0 + c4 * 4;
      if (i == 0) __builtin_prefetch(gp + 32, 0, 1);
      v4f d = *(const v4f*)gp;
      v4h h = { (_Float16)d[0], (_Float16)d[1], (_Float16)d[2], (_Float16)d[3] };
      *(v4h*)(Bs + r * LDA + c4 * 4) = h;
    }
    __syncthreads();
    v16h a[2], b[4];
#pragma unroll
    for (int mi = 0; mi < 2; ++mi) a[mi] = frag_ld(As + (wm*32 + mi*16 + ml) * LDA, kh);
#pragma unroll
    for (int ni = 0; ni < 4; ++ni) b[ni] = frag_ld(Bs + (wn*64 + ni*16 + ml) * LDA, kh);
#pragma unroll
    for (int mi = 0; mi < 2; ++mi)
#pragma unroll
      for (int ni = 0; ni < 4; ++ni) acc[mi][ni] = wmma16(a[mi], b[ni], acc[mi][ni]);
  }

#pragma unroll
  for (int mi = 0; mi < 2; ++mi)
#pragma unroll
    for (int ni = 0; ni < 4; ++ni) {
      int n = nbase + wn*64 + ni*16 + ml;
#pragma unroll
      for (int r = 0; r < 8; ++r) {
        int m = mbase + wm*32 + mi*16 + r + 8*kh;
        out[(size_t)m * DD + n] = acc[mi][ni][r];
      }
    }
}

// ============================================================
// launch
// ============================================================
extern "C" void kernel_launch(void* const* d_in, const int* in_sizes, int n_in,
                              void* d_out, int out_size, void* d_ws, size_t ws_size,
                              hipStream_t stream) {
  const float* x    = (const float*)d_in[0];   // [B,S,D]
  const float* wqkv = (const float*)d_in[1];   // [3D,D]
  const float* wout = (const float*)d_in[2];   // [D,D]
  float* out = (float*)d_out;                  // [B,S,D]

  char* ws = (char*)d_ws;
  const size_t MB = 1024ull * 1024ull;
  _Float16* qf    = (_Float16*)(ws);                 // 64 MB  [B,H,S,64]
  _Float16* kT    = (_Float16*)(ws + 64  * MB);      // 64 MB  [B,H,64,S]
  _Float16* vT    = (_Float16*)(ws + 128 * MB);      // 64 MB  [B,H,64,S]
  _Float16* attn  = (_Float16*)(ws + 192 * MB);      // 64 MB  [B,S,H,64]
  float*    kv    = (float*)   (ws + 256 * MB);      // 1 MB   [BH,64,64]
  _Float16* kvT   = (_Float16*)(ws + 257 * MB);      // 0.5 MB [BH,64,64] transposed
  float*    ksum  = (float*)   (ws + 258 * MB);      // 16 KB  [BH,64]
  float*    rnorm = (float*)   (ws + 259 * MB);      // 2 MB   [BH,S]

  la_init_kv     <<<1024, 256, 0, stream>>>(kv);
  la_qkv_gemm    <<<dim3(N3 / 128, MROWS / 128), 256, 0, stream>>>(x, wqkv, qf, kT, vT);
  la_kv_partial  <<<dim3(64, BH), 256, 0, stream>>>(kT, vT, kv);
  la_ksum        <<<(BH * HD) / 8, 256, 0, stream>>>(kT, ksum);
  la_kv_finalize <<<(BH * HD * HD) / 256, 256, 0, stream>>>(kv, kvT);
  la_norm        <<<(BH * SS) / 256, 256, 0, stream>>>(qf, ksum, rnorm);
  la_attn        <<<dim3(SS / 128, BH), 256, 0, stream>>>(qf, kvT, rnorm, attn);
  la_out_gemm    <<<dim3(DD / 128, MROWS / 128), 256, 0, stream>>>(attn, wout, out);
}
